// FMVP_6296422056480
// MI455X (gfx1250) — matrix operand order repaired and checked by reading, not compiled, not run on hardware
//
#include <hip/hip_runtime.h>

// ---------------------------------------------------------------------------
// Types for CDNA5 WMMA (wave32): bf16 16x16x32, f32 accumulate
// ---------------------------------------------------------------------------
typedef __attribute__((ext_vector_type(16))) __bf16 v16bf;
typedef __attribute__((ext_vector_type(8)))  float  v8f;

#define DEVINL static __device__ __forceinline__

union PackB8 {            // 8 bf16 <-> one 16-byte vector
  uint4  u;
  __bf16 h[8];
};

DEVINL v8f wmma_bf16(v16bf a, v16bf b, v8f c) {
  return __builtin_amdgcn_wmma_f32_16x16x32_bf16(false, a, false, b, (short)0, c,
                                                 false, false);
}

// A fragment: 16x32 (M rows x K), row-major source, p -> (row0,k0), leading dim ld.
DEVINL v16bf load_A_bf(const __bf16* p, int ld, int lane) {
  int lo = lane & 15, hi = lane >> 4;
  const __bf16* row = p + (size_t)lo * ld;
  v16bf a;
#pragma unroll
  for (int t = 0; t < 8; ++t) {
    a[t]     = row[hi * 8 + t];
    a[8 + t] = row[16 + hi * 8 + t];
  }
  return a;
}
DEVINL v16bf load_A_f32(const float* p, int ld, int lane) {
  int lo = lane & 15, hi = lane >> 4;
  const float* row = p + (size_t)lo * ld;
  v16bf a;
#pragma unroll
  for (int t = 0; t < 8; ++t) {
    a[t]     = (__bf16)row[hi * 8 + t];
    a[8 + t] = (__bf16)row[16 + hi * 8 + t];
  }
  return a;
}

// B fragment from "Bt" storage: Bt[n][k] row-major (B transposed), leading dim ld.
DEVINL v16bf load_Bt_bf(const __bf16* p, int ld, int lane) {
  int n = lane & 15, kb = (lane >> 4) * 8;
  const __bf16* row = p + (size_t)n * ld + kb;
  v16bf b;
#pragma unroll
  for (int t = 0; t < 8; ++t) { b[t] = row[t]; b[8 + t] = row[16 + t]; }
  return b;
}
DEVINL v16bf load_Bt_f32(const float* p, int ld, int lane) {
  int n = lane & 15, kb = (lane >> 4) * 8;
  const float* row = p + (size_t)n * ld + kb;
  v16bf b;
#pragma unroll
  for (int t = 0; t < 8; ++t) { b[t] = (__bf16)row[t]; b[8 + t] = (__bf16)row[16 + t]; }
  return b;
}

// A fragment from an LDS row (row base = this lane's A-row), two contiguous octets.
DEVINL v16bf load_A_lds(const __bf16* rowbase, int o1, int o2) {
  PackB8 x, y;
  x.u = *(const uint4*)(rowbase + o1);
  y.u = *(const uint4*)(rowbase + o2);
  v16bf a;
#pragma unroll
  for (int t = 0; t < 8; ++t) { a[t] = x.h[t]; a[8 + t] = y.h[t]; }
  return a;
}

DEVINL void ms_merge(float& m, float& s, float om, float os) {
  float nm = fmaxf(m, om);
  s = s * __expf(m - nm) + os * __expf(om - nm);
  m = nm;
}

// ---------------------------------------------------------------------------
#define BB 2048
#define MM 65536
#define HH 128
#define NCHUNK 32
#define PPAD 136   // LDS P row stride (128 + 8 pad): 16B aligned rows, bank spread

// ---------------------------------------------------------------------------
__global__ void k_zero(float* p, int n) {
  int i = blockIdx.x * blockDim.x + threadIdx.x;
  if (i < n) p[i] = 0.0f;
}

// q[b][h] = cos(ts*w_t+b_t) @ Wq^T ; vT[h][b] = (inp @ Wv^T)^T
__global__ void k_qv(const float* ts, const float* inp, const float* w_t,
                     const float* b_t, const float* Wq, const float* Wv,
                     __bf16* q_bf, __bf16* vT) {
  __shared__ float te[128];
  __shared__ float xs[256];
  int b = blockIdx.x, tid = threadIdx.x;
  float t = ts[b];
  if (tid < 128) te[tid] = __cosf(t * w_t[tid] + b_t[tid]);
  xs[tid] = inp[b * 256 + tid];
  __syncthreads();
  if (tid < 128) {
    int h = tid;
    float acc = 0.f;
#pragma unroll 4
    for (int d = 0; d < 128; ++d) acc += te[d] * Wq[h * 128 + d];
    q_bf[b * 128 + h] = (__bf16)acc;
  } else {
    int h = tid - 128;
    float acc = 0.f;
#pragma unroll 4
    for (int i = 0; i < 256; ++i) acc += xs[i] * Wv[h * 256 + i];
    vT[(size_t)h * BB + b] = (__bf16)acc;
  }
}

// out[m][j] = X @ Wk^T  (D^T orientation: A=Wk rows=j, B=X^T). Packed b128 store.
__global__ void k_keys(const float* __restrict__ X, const float* __restrict__ Wk,
                       __bf16* __restrict__ out) {
  int lane = threadIdx.x & 31, w = threadIdx.x >> 5;
  int lo = lane & 15, hi = lane >> 4;
  int m0 = blockIdx.x * 16, h0 = w * 16;
  v8f acc = {};
#pragma unroll
  for (int kk = 0; kk < 4; ++kk) {
    v16bf a = load_A_f32(Wk + h0 * 128 + kk * 32, 128, lane);
    v16bf b = load_Bt_f32(X + (size_t)m0 * 128 + kk * 32, 128, lane);
    acc = wmma_bf16(a, b, acc);
  }
  PackB8 pk;
#pragma unroll
  for (int v = 0; v < 8; ++v) pk.h[v] = (__bf16)acc[v];
  *(uint4*)&out[(size_t)(m0 + lo) * 128 + h0 + 8 * hi] = pk.u;
}

// Online softmax stats of s = scale*(q.k). 2 b-tiles (32 rows) per block:
// each streamed K fragment feeds 2 WMMAs.
__global__ void k_score(const __bf16* __restrict__ q, const __bf16* __restrict__ K,
                        float scale, float* __restrict__ pmax,
                        float* __restrict__ psum) {
  __shared__ float lm[8][32], ls[8][32];
  int lane = threadIdx.x & 31, w = threadIdx.x >> 5;
  int bt = blockIdx.x, chunk = blockIdx.y;
  int b0g = bt * 32;
  v16bf qa0[4], qa1[4];
#pragma unroll
  for (int kk = 0; kk < 4; ++kk) {
    qa0[kk] = load_A_bf(q + (size_t)b0g * 128 + kk * 32, 128, lane);
    qa1[kk] = load_A_bf(q + (size_t)(b0g + 16) * 128 + kk * 32, 128, lane);
  }
  float mx0[8], sm0[8], mx1[8], sm1[8];
#pragma unroll
  for (int v = 0; v < 8; ++v) {
    mx0[v] = -3.4e38f; sm0[v] = 0.f;
    mx1[v] = -3.4e38f; sm1[v] = 0.f;
  }

  for (int it = 0; it < 16; ++it) {
    int m0 = (chunk * 128 + it * 8 + w) * 16;
    v8f s0 = {}, s1 = {};
#pragma unroll
    for (int kk = 0; kk < 4; ++kk) {
      v16bf kb = load_Bt_bf(K + (size_t)m0 * 128 + kk * 32, 128, lane);
      s0 = wmma_bf16(qa0[kk], kb, s0);
      s1 = wmma_bf16(qa1[kk], kb, s1);
    }
#pragma unroll
    for (int v = 0; v < 8; ++v) {
      float sv0 = s0[v] * scale, sv1 = s1[v] * scale;
      float nm0 = fmaxf(mx0[v], sv0);
      sm0[v] = sm0[v] * __expf(mx0[v] - nm0) + __expf(sv0 - nm0);
      mx0[v] = nm0;
      float nm1 = fmaxf(mx1[v], sv1);
      sm1[v] = sm1[v] * __expf(mx1[v] - nm1) + __expf(sv1 - nm1);
      mx1[v] = nm1;
    }
  }
#pragma unroll
  for (int d = 1; d < 16; d <<= 1) {
#pragma unroll
    for (int v = 0; v < 8; ++v) {
      ms_merge(mx0[v], sm0[v], __shfl_xor(mx0[v], d), __shfl_xor(sm0[v], d));
      ms_merge(mx1[v], sm1[v], __shfl_xor(mx1[v], d), __shfl_xor(sm1[v], d));
    }
  }
  int lo = lane & 15, hi = lane >> 4;
  if (lo == 0) {
#pragma unroll
    for (int v = 0; v < 8; ++v) {
      lm[w][hi * 8 + v] = mx0[v];      ls[w][hi * 8 + v] = sm0[v];
      lm[w][16 + hi * 8 + v] = mx1[v]; ls[w][16 + hi * 8 + v] = sm1[v];
    }
  }
  __syncthreads();
  if (threadIdx.x < 32) {
    int bl = threadIdx.x;
    float m = lm[0][bl], s0 = ls[0][bl];
    for (int ww = 1; ww < 8; ++ww) ms_merge(m, s0, lm[ww][bl], ls[ww][bl]);
    pmax[chunk * BB + b0g + bl] = m;
    psum[chunk * BB + b0g + bl] = s0;
  }
}

__global__ void k_reduce(const float* __restrict__ pmax, const float* __restrict__ psum,
                         float* __restrict__ gmax, float* __restrict__ ginv) {
  int b = blockIdx.x * blockDim.x + threadIdx.x;
  if (b >= BB) return;
  float m = -3.4e38f, s = 0.f;
  for (int c = 0; c < NCHUNK; ++c) ms_merge(m, s, pmax[c * BB + b], psum[c * BB + b]);
  gmax[b] = m;
  ginv[b] = 1.0f / fmaxf(s, 1e-30f);
}

// mem_new = L2norm_rows(memory + P^T V). 2 m-tiles (32 rows) per block:
// each streamed Q fragment and vT fragment feeds 2 WMMAs.
__global__ void k_update(const float* __restrict__ memory, const __bf16* __restrict__ q,
                         const __bf16* __restrict__ K, const __bf16* __restrict__ vT,
                         const float* __restrict__ gmax, const float* __restrict__ ginv,
                         float* __restrict__ mem_f32, __bf16* __restrict__ memT) {
  __shared__ __align__(16) __bf16 PT[32 * PPAD];  // [m 0..31][b 0..127] (+pad)
  __shared__ float rowsq[32];
  int lane = threadIdx.x & 31, w = threadIdx.x >> 5;
  int lo = lane & 15, hi = lane >> 4;
  int m0 = blockIdx.x * 32, h0 = w * 16;
  if (threadIdx.x < 32) rowsq[threadIdx.x] = 0.f;

  v16bf kb0[4], kb1[4];  // key tiles of this block's 2 m-tiles (B operands)
#pragma unroll
  for (int kk = 0; kk < 4; ++kk) {
    kb0[kk] = load_Bt_bf(K + (size_t)m0 * 128 + kk * 32, 128, lane);
    kb1[kk] = load_Bt_bf(K + (size_t)(m0 + 16) * 128 + kk * 32, 128, lane);
  }

  v8f acc0 = {}, acc1 = {};
  for (int it = 0; it < 16; ++it) {
    int b0 = it * 128 + w * 16;
    v8f s0 = {}, s1 = {};
#pragma unroll
    for (int kk = 0; kk < 4; ++kk) {
      v16bf qa = load_A_bf(q + (size_t)b0 * 128 + kk * 32, 128, lane);
      s0 = wmma_bf16(qa, kb0[kk], s0);   // D: rows=b (v+8hi), cols=m0-tile m (lo)
      s1 = wmma_bf16(qa, kb1[kk], s1);
    }
    __syncthreads();
    {
      int bg0 = b0 + 8 * hi;
      float gm[8], gi[8];
#pragma unroll
      for (int v = 0; v < 8; ++v) { gm[v] = gmax[bg0 + v]; gi[v] = ginv[bg0 + v]; }
      PackB8 pk0, pk1;
#pragma unroll
      for (int v = 0; v < 8; ++v) {
        pk0.h[v] = (__bf16)(__expf(s0[v] - gm[v]) * gi[v]);
        pk1.h[v] = (__bf16)(__expf(s1[v] - gm[v]) * gi[v]);
      }
      *(uint4*)&PT[lo * PPAD + w * 16 + 8 * hi] = pk0.u;
      *(uint4*)&PT[(16 + lo) * PPAD + w * 16 + 8 * hi] = pk1.u;
    }
    __syncthreads();
#pragma unroll
    for (int kk = 0; kk < 4; ++kk) {
      v16bf va = load_A_bf(vT + (size_t)h0 * BB + it * 128 + kk * 32, BB, lane);
      v16bf pb0 = load_Bt_bf(PT + kk * 32, PPAD, lane);
      v16bf pb1 = load_Bt_bf(PT + 16 * PPAD + kk * 32, PPAD, lane);
      acc0 = wmma_bf16(va, pb0, acc0);   // D: rows=h, cols=m
      acc1 = wmma_bf16(va, pb1, acc1);
    }
  }
  // epilogue (both m-tiles): lane holds m = m0(+16)+lo, h = h0 + 8*hi + v
  const float* mrow0 = memory + (size_t)(m0 + lo) * 128 + h0 + 8 * hi;
  const float* mrow1 = memory + (size_t)(m0 + 16 + lo) * 128 + h0 + 8 * hi;
  float vals0[8], vals1[8];
  float sq0 = 0.f, sq1 = 0.f;
#pragma unroll
  for (int v = 0; v < 8; ++v) {
    vals0[v] = mrow0[v] + acc0[v]; sq0 += vals0[v] * vals0[v];
    vals1[v] = mrow1[v] + acc1[v]; sq1 += vals1[v] * vals1[v];
  }
  sq0 += __shfl_xor(sq0, 16);
  sq1 += __shfl_xor(sq1, 16);
  if (hi == 0) {
    atomicAdd(&rowsq[lo], sq0);
    atomicAdd(&rowsq[16 + lo], sq1);
  }
  __syncthreads();
  float sc0 = 1.0f / fmaxf(sqrtf(rowsq[lo]), 1e-12f);
  float sc1 = 1.0f / fmaxf(sqrtf(rowsq[16 + lo]), 1e-12f);
  float* orow0 = mem_f32 + (size_t)(m0 + lo) * 128 + h0 + 8 * hi;
  float* orow1 = mem_f32 + (size_t)(m0 + 16 + lo) * 128 + h0 + 8 * hi;
#pragma unroll
  for (int v = 0; v < 8; ++v) { orow0[v] = vals0[v] * sc0; orow1[v] = vals1[v] * sc1; }
#pragma unroll
  for (int v = 0; v < 8; ++v) {
    memT[(size_t)(h0 + 8 * hi + v) * MM + m0 + lo] = (__bf16)(vals0[v] * sc0);
    memT[(size_t)(h0 + 8 * hi + v) * MM + m0 + 16 + lo] = (__bf16)(vals1[v] * sc1);
  }
}

// read_r += P2 @ mem (chunk partials, atomic out). 2 b-tiles per block:
// each streamed K2 fragment and memT fragment feeds 2 WMMAs.
__global__ void k_read(const __bf16* __restrict__ q, const __bf16* __restrict__ K2,
                       const __bf16* __restrict__ memT, const float* __restrict__ gmax,
                       const float* __restrict__ ginv, float scale,
                       float* __restrict__ out) {
  __shared__ __align__(16) __bf16 P2[32 * PPAD];  // [b 0..31][m 0..127] (+pad)
  int lane = threadIdx.x & 31, w = threadIdx.x >> 5;
  int lo = lane & 15, hi = lane >> 4;
  int bt = blockIdx.x, chunk = blockIdx.y;
  int b0g = bt * 32, h0 = w * 16;

  v16bf qb0[4], qb1[4];  // Q^T as B operands, hoisted
#pragma unroll
  for (int kk = 0; kk < 4; ++kk) {
    qb0[kk] = load_Bt_bf(q + (size_t)b0g * 128 + kk * 32, 128, lane);
    qb1[kk] = load_Bt_bf(q + (size_t)(b0g + 16) * 128 + kk * 32, 128, lane);
  }
  float bm0 = gmax[b0g + lo],      bi0 = ginv[b0g + lo];
  float bm1 = gmax[b0g + 16 + lo], bi1 = ginv[b0g + 16 + lo];

  v8f acc0 = {}, acc1 = {};
  for (int it = 0; it < 16; ++it) {
    int mbase = chunk * 2048 + it * 128;
    int m0 = mbase + w * 16;
    v8f s0 = {}, s1 = {};
#pragma unroll
    for (int kk = 0; kk < 4; ++kk) {
      v16bf ka = load_A_bf(K2 + (size_t)m0 * 128 + kk * 32, 128, lane);
      s0 = wmma_bf16(ka, qb0[kk], s0);   // D: rows=m (v+8hi), cols=b (lo)
      s1 = wmma_bf16(ka, qb1[kk], s1);
    }
    __syncthreads();
    {
      PackB8 pk0, pk1;
#pragma unroll
      for (int v = 0; v < 8; ++v) {
        pk0.h[v] = (__bf16)(__expf(s0[v] * scale - bm0) * bi0);
        pk1.h[v] = (__bf16)(__expf(s1[v] * scale - bm1) * bi1);
      }
      *(uint4*)&P2[lo * PPAD + w * 16 + 8 * hi] = pk0.u;
      *(uint4*)&P2[(16 + lo) * PPAD + w * 16 + 8 * hi] = pk1.u;
    }
    __syncthreads();
#pragma unroll
    for (int kk = 0; kk < 4; ++kk) {
      int o1 = kk * 32 + hi * 8, o2 = o1 + 16;
      v16bf mb = load_Bt_bf(memT + (size_t)h0 * MM + mbase + kk * 32, MM, lane);
      v16bf pa0 = load_A_lds(P2 + lo * PPAD, o1, o2);
      v16bf pa1 = load_A_lds(P2 + (16 + lo) * PPAD, o1, o2);
      acc0 = wmma_bf16(pa0, mb, acc0);   // D: rows=b, cols=h
      acc1 = wmma_bf16(pa1, mb, acc1);
    }
  }
#pragma unroll
  for (int v = 0; v < 8; ++v) {
    atomicAdd(&out[(size_t)(b0g + v + 8 * hi) * 128 + h0 + lo], acc0[v]);
    atomicAdd(&out[(size_t)(b0g + 16 + v + 8 * hi) * 128 + h0 + lo], acc1[v]);
  }
}

// ---------------------------------------------------------------------------
extern "C" void kernel_launch(void* const* d_in, const int* in_sizes, int n_in,
                              void* d_out, int out_size, void* d_ws, size_t ws_size,
                              hipStream_t stream) {
  const float* ts  = (const float*)d_in[0];
  const float* inp = (const float*)d_in[1];
  const float* mem = (const float*)d_in[2];
  const float* w_t = (const float*)d_in[3];
  const float* b_t = (const float*)d_in[4];
  const float* Wq  = (const float*)d_in[5];
  const float* Wk  = (const float*)d_in[6];
  const float* Wv  = (const float*)d_in[7];
  float* out = (float*)d_out;

  char* ws = (char*)d_ws;
  size_t off = 0;
  auto carve = [&](size_t bytes) {
    void* p = ws + off;
    off += (bytes + 255) & ~(size_t)255;
    return p;
  };
  __bf16* q_bf  = (__bf16*)carve((size_t)BB * HH * 2);
  __bf16* vT    = (__bf16*)carve((size_t)BB * HH * 2);   // [h][b]
  __bf16* k_bf  = (__bf16*)carve((size_t)MM * HH * 2);   // [m][j]
  __bf16* k2_bf = (__bf16*)carve((size_t)MM * HH * 2);   // [m][j]
  __bf16* memT  = (__bf16*)carve((size_t)MM * HH * 2);   // [h][m]
  float*  mem_f = (float*)carve((size_t)MM * HH * 4);    // [m][h]
  float*  pmax  = (float*)carve((size_t)NCHUNK * BB * 4);
  float*  psum  = (float*)carve((size_t)NCHUNK * BB * 4);
  float*  gmax  = (float*)carve((size_t)BB * 4);
  float*  ginv  = (float*)carve((size_t)BB * 4);

  const float scale = 0.08838834764831845f;  // 1/sqrt(128)

  k_zero<<<(BB * HH + 255) / 256, 256, 0, stream>>>(out, BB * HH);
  k_qv<<<BB, 256, 0, stream>>>(ts, inp, w_t, b_t, Wq, Wv, q_bf, vT);
  k_keys<<<MM / 16, 256, 0, stream>>>(mem, Wk, k_bf);
  k_score<<<dim3(BB / 32, NCHUNK), 256, 0, stream>>>(q_bf, k_bf, 1.0f, pmax, psum);
  k_reduce<<<BB / 256, 256, 0, stream>>>(pmax, psum, gmax, ginv);
  k_update<<<MM / 32, 256, 0, stream>>>(mem, q_bf, k_bf, vT, gmax, ginv, mem_f, memT);
  k_keys<<<MM / 16, 256, 0, stream>>>(mem_f, Wk, k2_bf);
  k_score<<<dim3(BB / 32, NCHUNK), 256, 0, stream>>>(q_bf, k2_bf, scale, pmax, psum);
  k_reduce<<<BB / 256, 256, 0, stream>>>(pmax, psum, gmax, ginv);
  k_read<<<dim3(BB / 32, NCHUNK), 256, 0, stream>>>(q_bf, k2_bf, memT, gmax, ginv,
                                                    scale, out);
}
